// ProtectedMultiheadAttention_5368709120013
// MI455X (gfx1250) — matrix operand order repaired and checked by reading, not compiled, not run on hardware
//
#include <hip/hip_runtime.h>
#include <hip/hip_bf16.h>

// -------- CDNA5 WMMA / TDM types --------
typedef __attribute__((ext_vector_type(16))) __bf16 v16bf;
typedef __attribute__((ext_vector_type(8)))  float  v8f;
typedef __attribute__((ext_vector_type(4)))  unsigned int v4u;
typedef __attribute__((ext_vector_type(8)))  int v8i;
typedef __attribute__((ext_vector_type(4)))  int v4i;

#define L_SEQ 2048
#define BSZ   2
#define NHEAD 16
#define DHEAD 64
#define DMODEL 1024
#define NTOK  (L_SEQ * BSZ)   // 4096 tokens, t = i*BSZ + b

#if defined(__AMDGCN__) && __has_builtin(__builtin_amdgcn_tensor_load_to_lds) && \
    __has_builtin(__builtin_amdgcn_s_wait_tensorcnt)
#define USE_TDM 1
#else
#define USE_TDM 0
#endif

__device__ __forceinline__ v8f wmma_bf16(v16bf a, v16bf b, v8f c) {
  // D = A(16x32 bf16) * B(32x16 bf16) + C(16x16 f32)
  return __builtin_amdgcn_wmma_f32_16x16x32_bf16(false, a, false, b, (short)0, c,
                                                 false, false);
}

// ---- A fragment (16x32 bf16): lane<16 row r0+lane, K k0..k0+15;
//      lane>=16: K k0+16..k0+31.
__device__ __forceinline__ v16bf load_a_bf16(const __bf16* A, int lda, int r0, int k0, int lane) {
  int r  = r0 + (lane & 15);
  int kb = k0 + ((lane >> 4) << 4);
  const __bf16* p = A + (size_t)r * lda + kb;
  v16bf out;
#pragma unroll
  for (int u = 0; u < 16; ++u) out[u] = p[u];
  return out;
}

// ---- B fragment (32x16): source stored [N][K]; lane = column N,
//      lane<16: K = k0..k0+15, lane>=16: K = k0+16..k0+31. 32B-aligned vector load.
__device__ __forceinline__ v16bf load_b_bf16(const __bf16* B, int ldb, int n0, int k0, int lane) {
  int nn = n0 + (lane & 15);
  int kb = k0 + ((lane >> 4) << 4);
  return *(const v16bf*)(B + (size_t)nn * ldb + kb);
}

// Guarded variant (rows outside [0,nrows) -> zeros) for the rel-shift band.
__device__ __forceinline__ v16bf load_b_bf16_guard(const __bf16* B, int ldb, int n0, int k0,
                                                   int lane, int nrows) {
  int nn = n0 + (lane & 15);
  int kb = k0 + ((lane >> 4) << 4);
  v16bf out;
  if ((unsigned)nn < (unsigned)nrows) {
    const __bf16* p = B + (size_t)nn * ldb + kb;
#pragma unroll
    for (int u = 0; u < 16; ++u) out[u] = p[u];
  } else {
#pragma unroll
    for (int u = 0; u < 16; ++u) out[u] = (__bf16)0.0f;
  }
  return out;
}

// B fragment from a 64x64 LDS tile stored [N][K] (K contiguous).
__device__ __forceinline__ v16bf load_b_lds(const __bf16 (*T)[64], int n0, int k0, int lane) {
  int nn = n0 + (lane & 15);
  int kb = k0 + ((lane >> 4) << 4);
  return *(const v16bf*)&T[nn][kb];
}

// A fragment of P (probabilities) from LDS [16][64]
__device__ __forceinline__ v16bf load_a_lds(const __bf16 (*P)[64], int k0, int lane) {
  int r  = lane & 15;
  int kb = k0 + ((lane >> 4) << 4);
  v16bf out;
#pragma unroll
  for (int u = 0; u < 16; ++u) out[u] = P[r][kb + u];
  return out;
}

__device__ __forceinline__ float redmax16(float x) {
#pragma unroll
  for (int m = 1; m <= 8; m <<= 1) x = fmaxf(x, __shfl_xor(x, m, 32));
  return x;
}
__device__ __forceinline__ float redsum16(float x) {
#pragma unroll
  for (int m = 1; m <= 8; m <<= 1) x += __shfl_xor(x, m, 32);
  return x;
}

#if USE_TDM
// -------- Tensor Data Mover: 2D tile (bf16) global -> LDS --------
typedef __attribute__((address_space(3))) const __bf16* lds_cbf16_t;
__device__ __forceinline__ void tdm_load_2d(const __bf16* ldsPtr, const __bf16* gPtr,
                                            unsigned tile0, unsigned tile1,
                                            unsigned td0, unsigned td1, unsigned stride0) {
  unsigned lds = (unsigned)(unsigned long long)(lds_cbf16_t)ldsPtr;  // LDS byte offset
  unsigned long long ga = (unsigned long long)gPtr;
  v4u g0;
  g0[0] = 1u;                                                  // count=1, user mode
  g0[1] = lds;                                                 // lds_addr
  g0[2] = (unsigned)(ga & 0xFFFFFFFFu);                        // global_addr[31:0]
  g0[3] = (unsigned)((ga >> 32) & 0x01FFFFFFu) | 0x80000000u;  // addr[56:32] | type=2
  v8i g1;
  g1[0] = (int)0x00010000u;                                    // wg_mask=0, data_size=2B
  g1[1] = (int)((td0 & 0xFFFFu) << 16);                        // tensor_dim0[15:0]
  g1[2] = (int)((td0 >> 16) | ((td1 & 0xFFFFu) << 16));        // td0[31:16] | td1[15:0]
  g1[3] = (int)((td1 >> 16) | (tile0 << 16));                  // td1[31:16] | tile_dim0
  g1[4] = (int)tile1;                                          // tile_dim1, tile_dim2=0
  g1[5] = (int)stride0;                                        // tensor_dim0_stride[31:0]
  g1[6] = 0;
  g1[7] = 0;
  v4i g2 = {0, 0, 0, 0};
  v4i g3 = {0, 0, 0, 0};
#if __clang_major__ >= 23
  v8i g4 = {0, 0, 0, 0, 0, 0, 0, 0};
  __builtin_amdgcn_tensor_load_to_lds(g0, g1, g2, g3, g4, 0);
#else
  __builtin_amdgcn_tensor_load_to_lds(g0, g1, g2, g3, 0);
#endif
}
#endif

// =====================================================================
// K0: elementwise f32 -> bf16 (weights/activations converted once)
// =====================================================================
__global__ __launch_bounds__(256) void cvt_bf16_kernel(const float* __restrict__ src,
                                                       __bf16* __restrict__ dst, int n) {
  int i = (blockIdx.x * 256 + threadIdx.x) * 4;
  if (i + 3 < n) {
    float4 f = *(const float4*)(src + i);
    dst[i + 0] = (__bf16)f.x; dst[i + 1] = (__bf16)f.y;
    dst[i + 2] = (__bf16)f.z; dst[i + 3] = (__bf16)f.w;
  }
}

// ---- shared GEMM inner loop: 32(M) x 64(N) per wave, K step 32, bf16 ----
__device__ __forceinline__ void gemm_tile(const __bf16* __restrict__ A, int lda,
                                          const __bf16* __restrict__ B, int ldb,
                                          int m0, int n0, int K, int lane,
                                          v8f acc[2][4]) {
  for (int k0 = 0; k0 < K; k0 += 32) {
    if (k0 + 32 < K) __builtin_prefetch(B + (size_t)(n0 + (lane & 15)) * ldb + k0 + 32, 0, 0);
    v16bf bfr[4];
#pragma unroll
    for (int ns = 0; ns < 4; ++ns) bfr[ns] = load_b_bf16(B, ldb, n0 + ns * 16, k0, lane);
    v16bf a0 = load_a_bf16(A, lda, m0, k0, lane);
    v16bf a1 = load_a_bf16(A, lda, m0 + 16, k0, lane);
#pragma unroll
    for (int ns = 0; ns < 4; ++ns) {
      acc[0][ns] = wmma_bf16(a0, bfr[ns], acc[0][ns]);
      acc[1][ns] = wmma_bf16(a1, bfr[ns], acc[1][ns]);
    }
  }
}

// =====================================================================
// K1: w_heads = query @ qkv_w^T ; split per head, add biases.
//   Qw/Qr : [b][n][i][d]   Kb : [b][n][j][d]   Vt : [b][n][d][j]
// =====================================================================
__global__ __launch_bounds__(128) void qkv_proj_kernel(
    const __bf16* __restrict__ queryB, const __bf16* __restrict__ qkvW,
    const float* __restrict__ r_w_bias, const float* __restrict__ r_r_bias,
    __bf16* __restrict__ Qw, __bf16* __restrict__ Qr,
    __bf16* __restrict__ Kb, __bf16* __restrict__ Vt) {
  int lane = threadIdx.x & 31, wv = threadIdx.x >> 5;
  int m0 = blockIdx.x * 128 + wv * 32;  // token rows (t = i*2+b)
  int n0 = blockIdx.y * 64;             // output columns in [0,3072)
  v8f zero = {};
  v8f acc[2][4] = {{zero, zero, zero, zero}, {zero, zero, zero, zero}};
  gemm_tile(queryB, DMODEL, qkvW, DMODEL, m0, n0, DMODEL, lane, acc);

  int rbase = (lane >> 4) * 8;
#pragma unroll
  for (int mi = 0; mi < 2; ++mi) {
#pragma unroll
    for (int ns = 0; ns < 4; ++ns) {
      int col = n0 + ns * 16 + (lane & 15);
      int part = col >> 10, within = col & 1023;
      int h = within >> 6, d = within & 63;
#pragma unroll
      for (int v = 0; v < 8; ++v) {
        int t = m0 + mi * 16 + rbase + v;
        int i = t >> 1, b = t & 1;
        float val = acc[mi][ns][v];
        size_t hb = (size_t)(b * NHEAD + h);
        if (part == 0) {
          Qw[(hb * L_SEQ + i) * DHEAD + d] = (__bf16)(val + r_w_bias[h * DHEAD + d]);
          Qr[(hb * L_SEQ + i) * DHEAD + d] = (__bf16)(val + r_r_bias[h * DHEAD + d]);
        } else if (part == 1) {
          Kb[(hb * L_SEQ + i) * DHEAD + d] = (__bf16)val;
        } else {
          Vt[(hb * DHEAD + d) * L_SEQ + i] = (__bf16)val;
        }
      }
    }
  }
}

// =====================================================================
// K2: r_head_k = r @ r_w^T -> Rk[n][j][d]
// =====================================================================
__global__ __launch_bounds__(128) void r_proj_kernel(
    const __bf16* __restrict__ rB, const __bf16* __restrict__ rW, __bf16* __restrict__ Rk) {
  int lane = threadIdx.x & 31, wv = threadIdx.x >> 5;
  int m0 = blockIdx.x * 128 + wv * 32;
  int n0 = blockIdx.y * 64;
  v8f zero = {};
  v8f acc[2][4] = {{zero, zero, zero, zero}, {zero, zero, zero, zero}};
  gemm_tile(rB, DMODEL, rW, DMODEL, m0, n0, DMODEL, lane, acc);

  int rbase = (lane >> 4) * 8;
#pragma unroll
  for (int mi = 0; mi < 2; ++mi) {
#pragma unroll
    for (int ns = 0; ns < 4; ++ns) {
      int col = n0 + ns * 16 + (lane & 15);
      int h = col >> 6, d = col & 63;
#pragma unroll
      for (int v = 0; v < 8; ++v) {
        int row = m0 + mi * 16 + rbase + v;
        Rk[((size_t)h * L_SEQ + row) * DHEAD + d] = (__bf16)acc[mi][ns][v];
      }
    }
  }
}

// =====================================================================
// K3: fused rel-attention. Block = 4 waves, 64 q-rows of one (b, head).
// Double-buffered TDM: waves 0/1 issue tensor_load_to_lds for tile t+1,
// s_wait_tensorcnt 1 retires tile t (TDM is in-order per wave), barrier,
// then all waves compute on tile t while the DMA for t+1 is in flight.
// AC WMMA, BD band WMMA (5 subtiles cover the 79-col rel-shift window),
// online softmax, PV WMMA.
// =====================================================================
__global__ __launch_bounds__(128) void attn_kernel(
    const __bf16* __restrict__ Qw, const __bf16* __restrict__ Qr,
    const __bf16* __restrict__ Kb, const __bf16* __restrict__ Vt,
    const __bf16* __restrict__ Rk, __bf16* __restrict__ attnW) {
  __shared__ float  bandLds[4][16][80];   // per-wave BD_pre band (f32), 20 KB
  __shared__ __bf16 pLds[4][16][64];      // per-wave probabilities, 8 KB
#if USE_TDM
  __shared__ __bf16 kTile[2][64][64];     // double-buffered K tile, 16 KB
  __shared__ __bf16 vTile[2][64][64];     // double-buffered V^T tile, 16 KB
#endif

  int lane = threadIdx.x & 31, wv = threadIdx.x >> 5;
  int bn = blockIdx.y;
  int b = bn >> 4, n = bn & 15;
  int iw = blockIdx.x * 64 + wv * 16;     // this wave's q rows [iw, iw+16)
  int rbase = (lane >> 4) * 8;

  const __bf16* Qw_h = Qw + (size_t)(b * NHEAD + n) * L_SEQ * DHEAD;
  const __bf16* Qr_h = Qr + (size_t)(b * NHEAD + n) * L_SEQ * DHEAD;
  const __bf16* K_h  = Kb + (size_t)(b * NHEAD + n) * L_SEQ * DHEAD;
  const __bf16* V_h  = Vt + (size_t)(b * NHEAD + n) * DHEAD * L_SEQ;
  const __bf16* R_h  = Rk + (size_t)n * L_SEQ * DHEAD;

  v16bf aqw0 = load_a_bf16(Qw_h, DHEAD, iw, 0, lane);
  v16bf aqw1 = load_a_bf16(Qw_h, DHEAD, iw, 32, lane);
  v16bf aqr0 = load_a_bf16(Qr_h, DHEAD, iw, 0, lane);
  v16bf aqr1 = load_a_bf16(Qr_h, DHEAD, iw, 32, lane);

  v8f zero = {};
  v8f o[4] = {zero, zero, zero, zero};
  float mrow[8], lrow[8];
#pragma unroll
  for (int v = 0; v < 8; ++v) { mrow[v] = -1.0e30f; lrow[v] = 0.0f; }

  const float scale = 0.125f;  // 1/sqrt(64)

#if USE_TDM
  // Prime the pipeline: DMA tile 0 into buffer 0.
  if (wv == 0) tdm_load_2d(&kTile[0][0][0], K_h, 64, 64, 64, 64, DHEAD);
  if (wv == 1) tdm_load_2d(&vTile[0][0][0], V_h, 64, 64, 64, 64, L_SEQ);
#endif

  for (int j0 = 0; j0 < L_SEQ; j0 += 64) {
#if USE_TDM
    int cur = (j0 >> 6) & 1;
    bool has_next = (j0 + 64) < L_SEQ;
    if (wv == 0) {
      if (has_next) {
        tdm_load_2d(&kTile[cur ^ 1][0][0], K_h + (size_t)(j0 + 64) * DHEAD, 64, 64, 64, 64, DHEAD);
        __builtin_amdgcn_s_wait_tensorcnt(1);  // tile `cur` retired; `cur^1` in flight
      } else {
        __builtin_amdgcn_s_wait_tensorcnt(0);
      }
    }
    if (wv == 1) {
      if (has_next) {
        tdm_load_2d(&vTile[cur ^ 1][0][0], V_h + j0 + 64, 64, 64, 64, 64, L_SEQ);
        __builtin_amdgcn_s_wait_tensorcnt(1);
      } else {
        __builtin_amdgcn_s_wait_tensorcnt(0);
      }
    }
    __syncthreads();
#else
    if (j0 + 64 < L_SEQ) {
      __builtin_prefetch(K_h + (size_t)(j0 + 64) * DHEAD, 0, 0);   // global_prefetch_b8
      __builtin_prefetch(V_h + (size_t)(lane & 15) * L_SEQ + j0 + 64, 0, 0);
    }
#endif

    // ---- AC = Qw @ K^T (16x64 tile) ----
    v8f s[4];
#pragma unroll
    for (int ns = 0; ns < 4; ++ns) {
      v8f acc = {};
#if USE_TDM
      acc = wmma_bf16(aqw0, load_b_lds(kTile[cur], ns * 16, 0, lane), acc);
      acc = wmma_bf16(aqw1, load_b_lds(kTile[cur], ns * 16, 32, lane), acc);
#else
      acc = wmma_bf16(aqw0, load_b_bf16(K_h, DHEAD, j0 + ns * 16, 0, lane), acc);
      acc = wmma_bf16(aqw1, load_b_bf16(K_h, DHEAD, j0 + ns * 16, 32, lane), acc);
#endif
      s[ns] = acc;
    }

    // ---- BD_pre band: columns m = j - i + L - 1, window width 79 -> 5 subtiles ----
    int msta = j0 + (L_SEQ - 16) - iw;
#pragma unroll
    for (int s5 = 0; s5 < 5; ++s5) {
      v8f bd = {};
      bd = wmma_bf16(aqr0, load_b_bf16_guard(R_h, DHEAD, msta + s5 * 16, 0, lane, L_SEQ), bd);
      bd = wmma_bf16(aqr1, load_b_bf16_guard(R_h, DHEAD, msta + s5 * 16, 32, lane, L_SEQ), bd);
      int colb = s5 * 16 + (lane & 15);
#pragma unroll
      for (int v = 0; v < 8; ++v) bandLds[wv][rbase + v][colb] = bd[v];
    }

    // ---- rel-shift gather + scale: band col = jl + 15 - il in [0, 78] ----
    float pv[4][8];
#pragma unroll
    for (int ns = 0; ns < 4; ++ns) {
      int jl = ns * 16 + (lane & 15);
#pragma unroll
      for (int v = 0; v < 8; ++v) {
        int il = rbase + v;
        pv[ns][v] = (s[ns][v] + bandLds[wv][il][jl + 15 - il]) * scale;
      }
    }

    // ---- online softmax update ----
    float corr[8];
#pragma unroll
    for (int v = 0; v < 8; ++v) {
      float tm = fmaxf(fmaxf(pv[0][v], pv[1][v]), fmaxf(pv[2][v], pv[3][v]));
      tm = redmax16(tm);
      float mnew = fmaxf(mrow[v], tm);
      corr[v] = __expf(mrow[v] - mnew);
      mrow[v] = mnew;
    }
#pragma unroll
    for (int ns = 0; ns < 4; ++ns)
#pragma unroll
      for (int v = 0; v < 8; ++v) pv[ns][v] = __expf(pv[ns][v] - mrow[v]);
#pragma unroll
    for (int v = 0; v < 8; ++v) {
      float rs = redsum16(pv[0][v] + pv[1][v] + pv[2][v] + pv[3][v]);
      lrow[v] = lrow[v] * corr[v] + rs;
    }
#pragma unroll
    for (int ds = 0; ds < 4; ++ds)
#pragma unroll
      for (int v = 0; v < 8; ++v) o[ds][v] *= corr[v];

    // ---- stage P (bf16) and accumulate PV ----
#pragma unroll
    for (int ns = 0; ns < 4; ++ns) {
      int jl = ns * 16 + (lane & 15);
#pragma unroll
      for (int v = 0; v < 8; ++v) pLds[wv][rbase + v][jl] = (__bf16)pv[ns][v];
    }
    v16bf ap0 = load_a_lds(pLds[wv], 0, lane);
    v16bf ap1 = load_a_lds(pLds[wv], 32, lane);
#pragma unroll
    for (int ds = 0; ds < 4; ++ds) {
#if USE_TDM
      o[ds] = wmma_bf16(ap0, load_b_lds(vTile[cur], ds * 16, 0, lane), o[ds]);
      o[ds] = wmma_bf16(ap1, load_b_lds(vTile[cur], ds * 16, 32, lane), o[ds]);
#else
      {
        int d = ds * 16 + (lane & 15);
        const __bf16* vp0 = V_h + (size_t)d * L_SEQ + j0 + ((lane >> 4) << 4);
        o[ds] = wmma_bf16(ap0, *(const v16bf*)vp0, o[ds]);
        o[ds] = wmma_bf16(ap1, *(const v16bf*)(vp0 + 32), o[ds]);
      }
#endif
    }
#if USE_TDM
    __syncthreads();  // all waves done with tile `cur` before it is overwritten
#endif
  }

  // ---- normalize and emit attn (bf16) as [t][n*64+d] for the out GEMM ----
  float inv[8];
#pragma unroll
  for (int v = 0; v < 8; ++v) inv[v] = 1.0f / lrow[v];
#pragma unroll
  for (int ds = 0; ds < 4; ++ds) {
    int d = ds * 16 + (lane & 15);
#pragma unroll
    for (int v = 0; v < 8; ++v) {
      int il = rbase + v;
      int t = (iw + il) * BSZ + b;
      attnW[(size_t)t * DMODEL + n * DHEAD + d] = (__bf16)(o[ds][v] * inv[v]);
    }
  }
}

// =====================================================================
// K4: out = attn @ out_w^T + out_b  (f32 result)
// =====================================================================
__global__ __launch_bounds__(128) void out_proj_kernel(
    const __bf16* __restrict__ attnW, const __bf16* __restrict__ outW,
    const float* __restrict__ out_b, float* __restrict__ out) {
  int lane = threadIdx.x & 31, wv = threadIdx.x >> 5;
  int m0 = blockIdx.x * 128 + wv * 32;
  int n0 = blockIdx.y * 64;
  v8f zero = {};
  v8f acc[2][4] = {{zero, zero, zero, zero}, {zero, zero, zero, zero}};
  gemm_tile(attnW, DMODEL, outW, DMODEL, m0, n0, DMODEL, lane, acc);

  int rbase = (lane >> 4) * 8;
#pragma unroll
  for (int mi = 0; mi < 2; ++mi) {
#pragma unroll
    for (int ns = 0; ns < 4; ++ns) {
      int col = n0 + ns * 16 + (lane & 15);
      float bias = out_b[col];
#pragma unroll
      for (int v = 0; v < 8; ++v) {
        int t = m0 + mi * 16 + rbase + v;
        out[(size_t)t * DMODEL + col] = acc[mi][ns][v] + bias;
      }
    }
  }
}

// =====================================================================
extern "C" void kernel_launch(void* const* d_in, const int* in_sizes, int n_in,
                              void* d_out, int out_size, void* d_ws, size_t ws_size,
                              hipStream_t stream) {
  (void)in_sizes; (void)n_in; (void)out_size; (void)ws_size;
  const float* query    = (const float*)d_in[0];
  const float* r        = (const float*)d_in[1];
  const float* r_w_bias = (const float*)d_in[2];
  const float* r_r_bias = (const float*)d_in[3];
  const float* qkv_w    = (const float*)d_in[4];
  const float* r_w      = (const float*)d_in[5];
  const float* out_w    = (const float*)d_in[6];
  const float* out_b    = (const float*)d_in[7];
  float* out = (float*)d_out;

  // Workspace layout (bf16), MB-aligned offsets.
  char* ws = (char*)d_ws;
  const size_t MB = 1024 * 1024;
  __bf16* Qw     = (__bf16*)(ws + 0 * MB);    // 8 MB
  __bf16* Qr     = (__bf16*)(ws + 8 * MB);    // 8 MB
  __bf16* Kb     = (__bf16*)(ws + 16 * MB);   // 8 MB
  __bf16* Vt     = (__bf16*)(ws + 24 * MB);   // 8 MB
  __bf16* Rk     = (__bf16*)(ws + 32 * MB);   // 4 MB
  __bf16* attnW  = (__bf16*)(ws + 36 * MB);   // 8 MB
  __bf16* qkvWB  = (__bf16*)(ws + 44 * MB);   // 6 MB
  __bf16* rWB    = (__bf16*)(ws + 50 * MB);   // 2 MB
  __bf16* outWB  = (__bf16*)(ws + 52 * MB);   // 2 MB
  __bf16* queryB = (__bf16*)(ws + 54 * MB);   // 8 MB
  __bf16* rB     = (__bf16*)(ws + 62 * MB);   // 4 MB

  dim3 blk(128);
  // One-time bf16 conversion of activations and weights.
  cvt_bf16_kernel<<<dim3((NTOK * DMODEL) / 1024), 256, 0, stream>>>(query, queryB, NTOK * DMODEL);
  cvt_bf16_kernel<<<dim3((L_SEQ * DMODEL) / 1024), 256, 0, stream>>>(r, rB, L_SEQ * DMODEL);
  cvt_bf16_kernel<<<dim3((3 * DMODEL * DMODEL) / 1024), 256, 0, stream>>>(qkv_w, qkvWB,
                                                                          3 * DMODEL * DMODEL);
  cvt_bf16_kernel<<<dim3((DMODEL * DMODEL) / 1024), 256, 0, stream>>>(r_w, rWB, DMODEL * DMODEL);
  cvt_bf16_kernel<<<dim3((DMODEL * DMODEL) / 1024), 256, 0, stream>>>(out_w, outWB,
                                                                      DMODEL * DMODEL);

  qkv_proj_kernel<<<dim3(NTOK / 128, (3 * DMODEL) / 64), blk, 0, stream>>>(
      queryB, qkvWB, r_w_bias, r_r_bias, Qw, Qr, Kb, Vt);
  r_proj_kernel<<<dim3(L_SEQ / 128, DMODEL / 64), blk, 0, stream>>>(rB, rWB, Rk);
  attn_kernel<<<dim3(L_SEQ / 64, BSZ * NHEAD), blk, 0, stream>>>(
      Qw, Qr, Kb, Vt, Rk, attnW);
  out_proj_kernel<<<dim3(NTOK / 128, DMODEL / 64), blk, 0, stream>>>(
      attnW, outWB, out_b, out);
}
